// GraphNeuralNetwork_46737834115716
// MI455X (gfx1250) — matrix operand order repaired and checked by reading, not compiled, not run on hardware
//
#include <hip/hip_runtime.h>

// ---------------------------------------------------------------- types
typedef __attribute__((ext_vector_type(16))) __bf16 v16bf;
typedef __attribute__((ext_vector_type(8)))  __bf16 v8bf;
typedef __attribute__((ext_vector_type(8)))  float  v8f;

#define NN   100000   // nodes
#define EE   640000   // edges
#define GG   1000     // graphs
#define DIN  100
#define DD   128
#define EPSB 1e-5f

// ---------------------------------------------------------------- helpers
__device__ __forceinline__ unsigned short f32_to_bf16(float f) {
  unsigned u = __builtin_bit_cast(unsigned, f);
  u += 0x7FFFu + ((u >> 16) & 1u);     // round-to-nearest-even
  return (unsigned short)(u >> 16);
}
__device__ __forceinline__ float bf16_to_f32(unsigned short h) {
  unsigned u = ((unsigned)h) << 16;
  return __builtin_bit_cast(float, u);
}
// CDNA5 hardware f32 atomic add (no-return form -> STOREcnt; s_endpgm waits idle)
__device__ __forceinline__ void atomAddF(float* p, float v) {
  asm volatile("global_atomic_add_f32 %0, %1, off" : : "v"(p), "v"(v) : "memory");
}

// A-fragment: 8 contiguous halves at p, 8 contiguous halves at p+16
__device__ __forceinline__ v16bf ld_frag_gap16(const unsigned short* p) {
  v8bf a = *(const v8bf*)p;
  v8bf b = *(const v8bf*)(p + 16);
  return __builtin_shufflevector(a, b, 0,1,2,3,4,5,6,7,8,9,10,11,12,13,14,15);
}
// B-fragment: 16 contiguous halves at p
__device__ __forceinline__ v16bf ld_frag_contig(const unsigned short* p) {
  v8bf a = *(const v8bf*)p;
  v8bf b = *(const v8bf*)(p + 8);
  return __builtin_shufflevector(a, b, 0,1,2,3,4,5,6,7,8,9,10,11,12,13,14,15);
}
__device__ __forceinline__ v8f wmma_bf16(v16bf a, v16bf b, v8f c) {
  return __builtin_amdgcn_wmma_f32_16x16x32_bf16(false, a, false, b, (short)0, c,
                                                 false, false);
}

// ---------------------------------------------------------------- degree / norm
__global__ void k_deg_init(float* deg) {
  int i = blockIdx.x * 256 + threadIdx.x;
  if (i < NN) deg[i] = 1.0f;                     // self-loop weight
}
__global__ void k_deg_acc(float* deg, const int* __restrict__ dst,
                          const float* __restrict__ ew) {
  int e = blockIdx.x * 256 + threadIdx.x;
  if (e < EE) atomAddF(&deg[dst[e]], ew[e]);
}
__global__ void k_dinv(float* dinv /*in: deg*/, float* nself) {
  int i = blockIdx.x * 256 + threadIdx.x;
  if (i < NN) {
    float d  = dinv[i];                          // >= 1 (self loop)
    float di = rsqrtf(d);
    dinv[i]  = di;
    nself[i] = di * di;                          // self-loop edge norm
  }
}
__global__ void k_norm(float* norm, const float* __restrict__ dinv,
                       const int* __restrict__ src, const int* __restrict__ dst,
                       const float* __restrict__ ew) {
  int e = blockIdx.x * 256 + threadIdx.x;
  if (e < EE) norm[e] = dinv[src[e]] * ew[e] * dinv[dst[e]];
}

// ---------------------------------------------------------------- GEMM (bf16x2 WMMA)
// hw[M x 128] = A[M x K] * W[K x 128], K <= 128 (zero padded)
// fused epilogue: agg[i][n] = hw[i][n] * nself[i]   (self-loop init of scatter sum)
#define LDK 72   // LDS K-stride (halves): 144B rows -> 36-bank rotation
__global__ __launch_bounds__(256) void k_gemm(const float* __restrict__ A,
                                              const float* __restrict__ W,
                                              const float* __restrict__ nself,
                                              float* __restrict__ hw,
                                              float* __restrict__ agg,
                                              int M, int K) {
  __shared__ __align__(16) unsigned short sAh[64 * LDK];
  __shared__ __align__(16) unsigned short sAl[64 * LDK];
  __shared__ __align__(16) unsigned short sBh[128 * LDK];
  __shared__ __align__(16) unsigned short sBl[128 * LDK];

  const int tid    = threadIdx.x;
  const int lane   = tid & 31;
  const int wave   = tid >> 5;
  const int blockM = blockIdx.x * 64;

  const int rowA    = (wave & 3) * 16 + (lane & 15);  // A fragment row (0..63)
  const int hi      = lane >> 4;                      // lane half
  const int colBase = (wave >> 2) * 64;               // this wave's 64 columns

  v8f acc[4];
#pragma unroll
  for (int i = 0; i < 4; ++i)
    acc[i] = (v8f){0.f, 0.f, 0.f, 0.f, 0.f, 0.f, 0.f, 0.f};

#pragma unroll
  for (int kc = 0; kc < 2; ++kc) {                    // two 64-wide K chunks
    const int kb0 = kc * 64;
    // stage A chunk (64 x 64), hi/lo bf16 split, zero pad
    for (int idx = tid; idx < 64 * 64; idx += 256) {
      int r = idx >> 6, kl = idx & 63;
      int g = blockM + r, k = kb0 + kl;
      float v = (g < M && k < K) ? A[(size_t)g * K + k] : 0.0f;
      unsigned short h = f32_to_bf16(v);
      sAh[r * LDK + kl] = h;
      sAl[r * LDK + kl] = f32_to_bf16(v - bf16_to_f32(h));
    }
    // stage W chunk transposed (n-major: sB[n][k]) so B frags are contiguous
    for (int idx = tid; idx < 128 * 64; idx += 256) {
      int n = idx >> 6, kl = idx & 63;
      int k = kb0 + kl;
      float v = (k < K) ? W[k * DD + n] : 0.0f;
      unsigned short h = f32_to_bf16(v);
      sBh[n * LDK + kl] = h;
      sBl[n * LDK + kl] = f32_to_bf16(v - bf16_to_f32(h));
    }
    __syncthreads();

#pragma unroll
    for (int kk = 0; kk < 2; ++kk) {                  // k step 32 within chunk
      const int ka = kk * 32 + hi * 16;               // A: K 0..15 / 16..31 per half
      v16bf ah = ld_frag_contig(&sAh[rowA * LDK + ka]);
      v16bf al = ld_frag_contig(&sAl[rowA * LDK + ka]);
      const int kb = kk * 32 + hi * 16;               // B: K 0..15 / 16..31 per half
#pragma unroll
      for (int nt = 0; nt < 4; ++nt) {
        int n = colBase + nt * 16 + (lane & 15);
        v16bf bh = ld_frag_contig(&sBh[n * LDK + kb]);
        v16bf bl = ld_frag_contig(&sBl[n * LDK + kb]);
        acc[nt] = wmma_bf16(ah, bh, acc[nt]);         // hi*hi
        acc[nt] = wmma_bf16(ah, bl, acc[nt]);         // hi*lo
        acc[nt] = wmma_bf16(al, bh, acc[nt]);         // lo*hi
      }
    }
    __syncthreads();
  }

  // store D: VGPR r -> row (hi*8 + r), lane&15 -> col
  const int rbase = blockM + (wave & 3) * 16 + hi * 8;
  float ns[8];
#pragma unroll
  for (int r = 0; r < 8; ++r)
    ns[r] = (rbase + r < M) ? nself[rbase + r] : 0.0f;
#pragma unroll
  for (int nt = 0; nt < 4; ++nt) {
    int col = colBase + nt * 16 + (lane & 15);
#pragma unroll
    for (int r = 0; r < 8; ++r) {
      int grow = rbase + r;
      if (grow < M) {
        float v = acc[nt][r];
        hw [(size_t)grow * DD + col] = v;
        agg[(size_t)grow * DD + col] = v * ns[r];     // fused self-loop init
      }
    }
  }
}

// ---------------------------------------------------------------- aggregation
__global__ void k_scatter(const float* __restrict__ hw, float* __restrict__ agg,
                          const int* __restrict__ src, const int* __restrict__ dst,
                          const float* __restrict__ norm) {
  int e = blockIdx.x * 8 + (threadIdx.x >> 5);       // one wave per edge
  if (e >= EE) return;
  int lane = threadIdx.x & 31;
  int s = src[e], d = dst[e];
  float w = norm[e];
  const float4 v = ((const float4*)(hw + (size_t)s * DD))[lane];
  float* p = agg + (size_t)d * DD + lane * 4;
  atomAddF(p + 0, v.x * w);
  atomAddF(p + 1, v.y * w);
  atomAddF(p + 2, v.z * w);
  atomAddF(p + 3, v.w * w);
}
__global__ void k_bn_relu(const float* __restrict__ agg, const float* __restrict__ b,
                          const float* __restrict__ g, const float* __restrict__ bt,
                          const float* __restrict__ rm, const float* __restrict__ rv,
                          float* __restrict__ h) {
  int idx = blockIdx.x * 256 + threadIdx.x;          // < NN*DD exactly
  int d = idx & (DD - 1);
  float v = (agg[idx] + b[d] - rm[d]) * rsqrtf(rv[d] + EPSB) * g[d] + bt[d];
  h[idx] = fmaxf(v, 0.0f);
}

// ---------------------------------------------------------------- pooling
__global__ void k_pool_zero(float* pooled, float* counts) {
  int idx = blockIdx.x * 256 + threadIdx.x;
  if (idx < GG * DD) pooled[idx] = 0.0f;
  else if (idx < GG * DD + GG) counts[idx - GG * DD] = 0.0f;
}
__global__ void k_pool_acc(const float* __restrict__ h, const int* __restrict__ batch,
                           float* pooled, float* counts) {
  int i = blockIdx.x * 8 + (threadIdx.x >> 5);       // one wave per node
  if (i >= NN) return;
  int lane = threadIdx.x & 31;
  int g = batch[i];
  const float4 v = ((const float4*)(h + (size_t)i * DD))[lane];
  float* p = pooled + (size_t)g * DD + lane * 4;
  atomAddF(p + 0, v.x);
  atomAddF(p + 1, v.y);
  atomAddF(p + 2, v.z);
  atomAddF(p + 3, v.w);
  if (lane == 0) atomAddF(&counts[g], 1.0f);
}
__global__ void k_pool_out(const float* __restrict__ pooled,
                           const float* __restrict__ counts,
                           const float* __restrict__ Wout,
                           const float* __restrict__ bout, float* out) {
  __shared__ float red[DD];
  int g = blockIdx.x, t = threadIdx.x;
  red[t] = pooled[g * DD + t] * Wout[t];
  __syncthreads();
#pragma unroll
  for (int s = 64; s > 0; s >>= 1) {
    if (t < s) red[t] += red[t + s];
    __syncthreads();
  }
  if (t == 0) out[g] = red[0] / fmaxf(counts[g], 1.0f) + bout[0];
}

// ---------------------------------------------------------------- launch
extern "C" void kernel_launch(void* const* d_in, const int* in_sizes, int n_in,
                              void* d_out, int out_size, void* d_ws, size_t ws_size,
                              hipStream_t stream) {
  const float* x     = (const float*)d_in[0];
  const int*   eidx  = (const int*)d_in[1];
  const float* ew    = (const float*)d_in[2];
  const int*   batch = (const int*)d_in[3];
  const float* W[3]  = {(const float*)d_in[4], (const float*)d_in[6], (const float*)d_in[8]};
  const float* B[3]  = {(const float*)d_in[5], (const float*)d_in[7], (const float*)d_in[9]};
  const float* gamma = (const float*)d_in[10];
  const float* beta  = (const float*)d_in[11];
  const float* rm    = (const float*)d_in[12];
  const float* rv    = (const float*)d_in[13];
  const float* Wout  = (const float*)d_in[14];
  const float* bout  = (const float*)d_in[15];
  float* out = (float*)d_out;

  const int* src = eidx;
  const int* dst = eidx + EE;

  // workspace carve-up (all offsets 256B aligned)
  char* base = (char*)d_ws;
  const size_t NB = (size_t)NN * DD * sizeof(float);       // 51.2 MB
  float* hw     = (float*)(base);                          // GEMM output
  float* agg    = (float*)(base + NB);                     // scatter target
  float* h      = (float*)(base + 2 * NB);                 // layer activations
  char*  p      = base + 3 * NB;
  float* dinv   = (float*)p;  p += ((size_t)NN * 4 + 255) & ~255ull;
  float* nself  = (float*)p;  p += ((size_t)NN * 4 + 255) & ~255ull;
  float* norm   = (float*)p;  p += ((size_t)EE * 4 + 255) & ~255ull;
  float* pooled = (float*)p;  p += ((size_t)GG * DD * 4 + 255) & ~255ull;
  float* counts = (float*)p;
  (void)ws_size; (void)n_in; (void)in_sizes; (void)out_size;

  // ---- normalization precompute
  k_deg_init<<<(NN + 255) / 256, 256, 0, stream>>>(dinv);
  k_deg_acc <<<(EE + 255) / 256, 256, 0, stream>>>(dinv, dst, ew);
  k_dinv    <<<(NN + 255) / 256, 256, 0, stream>>>(dinv, nself);
  k_norm    <<<(EE + 255) / 256, 256, 0, stream>>>(norm, dinv, src, dst, ew);

  // ---- 3 GCN layers
  const float* in = x;
  int K = DIN;
  const int ND_BLOCKS = (NN * DD) / 256;          // 50000 exactly
  for (int l = 0; l < 3; ++l) {
    k_gemm    <<<(NN + 63) / 64, 256, 0, stream>>>(in, W[l], nself, hw, agg, NN, K);
    k_scatter <<<EE / 8, 256, 0, stream>>>(hw, agg, src, dst, norm);
    k_bn_relu <<<ND_BLOCKS, 256, 0, stream>>>(agg, B[l], gamma + l * DD, beta + l * DD,
                                              rm + l * DD, rv + l * DD, h);
    in = h;
    K = DD;
  }

  // ---- global mean pool + output head
  k_pool_zero<<<(GG * DD + GG + 255) / 256, 256, 0, stream>>>(pooled, counts);
  k_pool_acc <<<NN / 8, 256, 0, stream>>>(h, batch, pooled, counts);
  k_pool_out <<<GG, DD, 0, stream>>>(pooled, counts, Wout, bout, out);
}
